// CausalMHA_27376121545423
// MI455X (gfx1250) — matrix operand order repaired and checked by reading, not compiled, not run on hardware
//
#include <hip/hip_runtime.h>

typedef __attribute__((ext_vector_type(16))) __bf16 v16bf;
typedef __attribute__((ext_vector_type(8)))  __bf16 v8bf;
typedef __attribute__((ext_vector_type(8)))  float  v8f;
typedef __attribute__((ext_vector_type(4)))  unsigned int v4u;
typedef __attribute__((ext_vector_type(8)))  int v8i_;
typedef __attribute__((ext_vector_type(4)))  int v4i_;

#define B_    2
#define S_    2048
#define D_    2048
#define H_    16
#define HD_   128
#define NQKV_ 6144
#define MROWS_ 4096   // B_*S_

#if __has_builtin(__builtin_amdgcn_tensor_load_to_lds) && \
    __has_builtin(__builtin_amdgcn_s_wait_tensorcnt)
#define USE_TDM 1
#else
#define USE_TDM 0
#endif

__device__ __forceinline__ unsigned short f2bf(float f) {
  unsigned u = __float_as_uint(f);
  u += 0x7FFFu + ((u >> 16) & 1u);           // round-to-nearest-even
  return (unsigned short)(u >> 16);
}

// ---------------- fp32 -> bf16 convert ----------------
__global__ void cvt_bf16_kernel(const float* __restrict__ in,
                                unsigned short* __restrict__ out, int n) {
  int i = blockIdx.x * blockDim.x + threadIdx.x;
  int stride = gridDim.x * blockDim.x;
  for (; i < n; i += stride) out[i] = f2bf(in[i]);
}

// ---------------- fragment loaders ----------------
// A fragment 16x32 (MxK), row-major source. lanes 0-15: K {0..7,16..23},
// lanes 16-31: K {8..15,24..31} (ISA 7.12.2, 16-bit A 16x32).
__device__ __forceinline__ v16bf load_a_frag(const unsigned short* A, int lda,
                                             int row0, int k0, int lane) {
  int r = lane & 15, hl = lane >> 4;
  const unsigned short* p = A + (size_t)(row0 + r) * lda + k0 + hl * 8;
  union { v16bf v; v8bf h[2]; } u;
  u.h[0] = *(const v8bf*)p;
  u.h[1] = *(const v8bf*)(p + 16);
  return u.v;
}

// B fragment 32x16 (KxN): lane l = column N=l%16; lanes 0-15: K {0..7,16..23},
// lanes 16-31: K {8..15,24..31}.
__device__ __forceinline__ v16bf load_b_frag(const unsigned short* Bm, int ldb,
                                             int k0, int n0, int lane) {
  const unsigned short* p = Bm + (size_t)(k0 + 8 * (lane >> 4)) * ldb + n0 + (lane & 15);
  v16bf v;
  #pragma unroll
  for (int s = 0; s < 16; ++s) {
    unsigned short x = p[(size_t)(s < 8 ? s : s + 8) * ldb];
    v[s] = __builtin_bit_cast(__bf16, x);
  }
  return v;
}

// B fragment straight from a contiguous row pointer (LDS tile).
__device__ __forceinline__ v16bf load_b_frag_ptr(const unsigned short* p) {
  v16bf v;
  #pragma unroll
  for (int s = 0; s < 16; ++s) {
    unsigned short x = p[(s < 8 ? s : s + 8) * 512];
    v[s] = __builtin_bit_cast(__bf16, x);
  }
  return v;
}

#if USE_TDM
// Post a TDM descriptor: load a 32-row x 512-col bf16 tile of W (row stride
// ldb elements) into LDS at byte offset lds_off.  ISA 8.3/8.4 bit packing.
__device__ __forceinline__ void tdm_load_w_tile(const unsigned short* gsrc,
                                                unsigned lds_off, int ldb) {
  unsigned long long ga = (unsigned long long)(const void*)gsrc;
  v4u g0;
  g0.x = 1u;                                  // count=1, user mode, no gather
  g0.y = lds_off;                             // lds_addr (bytes)
  g0.z = (unsigned)ga;                        // global_addr[31:0]
  g0.w = (unsigned)(ga >> 32) | (2u << 30);   // global_addr[56:32] | type=2
  v8i_ g1;
  g1[0] = 0x10000;                            // data_size=2B, no pad/iterate
  g1[1] = (int)(512u << 16);                  // tensor_dim0 = 512 (lo16)
  g1[2] = (int)(32u << 16);                   // dim0 hi16=0 | tensor_dim1=32
  g1[3] = (int)(512u << 16);                  // dim1 hi16=0 | tile_dim0=512
  g1[4] = 32;                                 // tile_dim1=32, tile_dim2=0
  g1[5] = ldb;                                // tensor_dim0_stride (elements)
  g1[6] = 0;
  g1[7] = 0;
  v4i_ z4 = {0, 0, 0, 0};
  v8i_ z8 = {0, 0, 0, 0, 0, 0, 0, 0};
  __builtin_amdgcn_tensor_load_to_lds(g0, g1, z4, z4, z8, 0);
}
#endif

// ---------------- GEMM: block = 8 waves, tile 16(M) x 512(N), k-step 32 ----
// TDM double-buffers the 32x512 W tile in LDS; waves read B-frags via DS.
__device__ __forceinline__ void gemm_core(const unsigned short* __restrict__ A, int lda,
                                          const unsigned short* __restrict__ W, int ldb,
                                          int K, int row0, int nblk0,
                                          unsigned short* wbuf /* [2][32][512] LDS */,
                                          int lane, int wave, v8f acc[4]) {
  #pragma unroll
  for (int j = 0; j < 4; ++j)
    #pragma unroll
    for (int e = 0; e < 8; ++e) acc[j][e] = 0.0f;

#if USE_TDM
  if (wave == 0) tdm_load_w_tile(W + nblk0, 0u, ldb);
  int p = 0;
  for (int k0 = 0; k0 < K; k0 += 32, p ^= 1) {
    if (wave == 0) {
      if (k0 + 32 < K) {
        tdm_load_w_tile(W + (size_t)(k0 + 32) * ldb + nblk0,
                        (unsigned)((p ^ 1) * 32768), ldb);
        __builtin_amdgcn_s_wait_tensorcnt(1);   // current buffer complete
      } else {
        __builtin_amdgcn_s_wait_tensorcnt(0);
      }
    }
    __syncthreads();
    // The TDM engine wrote wbuf behind the compiler's back: force it to treat
    // the LDS tile as modified so the B-fragment DS loads are not folded away.
    asm volatile("" : : "v"(wbuf) : "memory");
    v16bf a = load_a_frag(A, lda, row0, k0, lane);
    if (k0 + 32 < K)
      __builtin_prefetch(A + (size_t)(row0 + (lane & 15)) * lda + k0 + 32, 0, 1);
    const unsigned short* brow = wbuf + (size_t)p * 32 * 512
                               + (size_t)(8 * (lane >> 4)) * 512 + wave * 64 + (lane & 15);
    #pragma unroll
    for (int j = 0; j < 4; ++j) {
      v16bf bf = load_b_frag_ptr(brow + j * 16);
      acc[j] = __builtin_amdgcn_wmma_f32_16x16x32_bf16(
          false, a, false, bf, (short)0, acc[j], false, false);
    }
    __syncthreads();
  }
#else
  (void)wbuf;
  int n0 = nblk0 + wave * 64;
  for (int k0 = 0; k0 < K; k0 += 32) {
    v16bf a = load_a_frag(A, lda, row0, k0, lane);
    if (k0 + 32 < K)
      __builtin_prefetch(W + (size_t)(k0 + 32 + lane) * ldb + n0, 0, 1);
    #pragma unroll
    for (int j = 0; j < 4; ++j) {
      v16bf bf = load_b_frag(W, ldb, k0, n0 + j * 16, lane);
      acc[j] = __builtin_amdgcn_wmma_f32_16x16x32_bf16(
          false, a, false, bf, (short)0, acc[j], false, false);
    }
  }
#endif
}

// ---------------- QKV projection: X[4096][2048] @ W[2048][6144] ----------------
__global__ void __launch_bounds__(256) gemm_qkv_kernel(
    const unsigned short* __restrict__ X,
    const unsigned short* __restrict__ W,
    unsigned short* __restrict__ Qb,
    unsigned short* __restrict__ Ktb,
    unsigned short* __restrict__ Vb) {
  __shared__ __align__(16) unsigned short wtile[2][32][512];  // 64 KB, LDS off 0
  int lane = threadIdx.x & 31;
  int wave = threadIdx.x >> 5;
  int nblk0 = blockIdx.x * 512;
  int row0  = blockIdx.y * 16;

  v8f acc[4];
  gemm_core(X, D_, W, NQKV_, D_, row0, nblk0, &wtile[0][0][0], lane, wave, acc);

  int hl = lane >> 4, nn = lane & 15;
  int r0 = row0 + 8 * hl;
  int bb = row0 >> 11;         // which batch (2048 rows per batch)
  int s0 = r0 & (S_ - 1);
  int n0 = nblk0 + wave * 64;

  #pragma unroll
  for (int j = 0; j < 4; ++j) {
    int c0 = n0 + j * 16;
    int which = c0 >> 11;               // 0=Q 1=K 2=V
    int hh = (c0 >> 7) & (H_ - 1);
    int d0 = (c0 & (HD_ - 1)) + nn;
    if (which == 1) {                   // K -> transposed [b,h,d,s]
      unsigned short* dst = Ktb + ((size_t)((bb * H_ + hh) * HD_) + d0) * S_ + s0;
      #pragma unroll
      for (int i = 0; i < 8; ++i) dst[i] = f2bf(acc[j][i]);
    } else {
      unsigned short* dst = (which == 0 ? Qb : Vb)
                          + ((size_t)(bb * H_ + hh) * S_ + s0) * HD_ + d0;
      #pragma unroll
      for (int i = 0; i < 8; ++i) dst[(size_t)i * HD_] = f2bf(acc[j][i]);
    }
  }
}

// ---------------- flash attention: one 16-query tile per wave ----------------
__global__ void __launch_bounds__(128) attn_kernel(
    const unsigned short* __restrict__ Qb,
    const unsigned short* __restrict__ Ktb,
    const unsigned short* __restrict__ Vb,
    unsigned short* __restrict__ Ob) {
  __shared__ __align__(16) unsigned short pls[4][16][32];  // P transpose buffer
  int lane = threadIdx.x & 31;
  int wave = threadIdx.x >> 5;
  int hl = lane >> 4, nn = lane & 15;
  int hh = blockIdx.y, bb = blockIdx.z;
  int qbase = blockIdx.x * 64 + wave * 16;

  const unsigned short* Qp  = Qb  + (size_t)(bb * H_ + hh) * S_ * HD_;
  const unsigned short* Ktp = Ktb + (size_t)(bb * H_ + hh) * HD_ * S_;
  const unsigned short* Vp  = Vb  + (size_t)(bb * H_ + hh) * S_ * HD_;

  v16bf qf[4];
  #pragma unroll
  for (int kk = 0; kk < 4; ++kk) qf[kk] = load_a_frag(Qp, HD_, qbase, kk * 32, lane);

  v8f acc[8];
  #pragma unroll
  for (int t = 0; t < 8; ++t)
    #pragma unroll
    for (int e = 0; e < 8; ++e) acc[t][e] = 0.0f;
  float mrun[8], lrun[8];
  #pragma unroll
  for (int i = 0; i < 8; ++i) { mrun[i] = -1e30f; lrun[i] = 0.0f; }

  const float scale = 0.08838834764831845f;  // 1/sqrt(128)

  for (int kv0 = 0; kv0 < qbase + 16; kv0 += 32) {
    // ---- S = Q K^T : two 16x16 score tiles (keys kv0..kv0+31) ----
    v8f sc[2];
    #pragma unroll
    for (int t = 0; t < 2; ++t) {
      #pragma unroll
      for (int e = 0; e < 8; ++e) sc[t][e] = 0.0f;
      int key0 = kv0 + t * 16;
      #pragma unroll
      for (int kk = 0; kk < 4; ++kk) {
        v16bf kb = load_b_frag(Ktp, S_, kk * 32, key0, lane);
        sc[t] = __builtin_amdgcn_wmma_f32_16x16x32_bf16(
            false, qf[kk], false, kb, (short)0, sc[t], false, false);
      }
    }
    // ---- scale + causal mask ----
    #pragma unroll
    for (int t = 0; t < 2; ++t) {
      int key = kv0 + t * 16 + nn;
      #pragma unroll
      for (int i = 0; i < 8; ++i) {
        int qq = qbase + i + 8 * hl;
        float v = sc[t][i] * scale;
        sc[t][i] = (key <= qq) ? v : -1e30f;
      }
    }
    // ---- online softmax (rows live across the 16-lane halves) ----
    float mnew[8], scl[8];
    #pragma unroll
    for (int i = 0; i < 8; ++i) {
      float v = fmaxf(sc[0][i], sc[1][i]);
      v = fmaxf(v, __shfl_xor(v, 1, 32));
      v = fmaxf(v, __shfl_xor(v, 2, 32));
      v = fmaxf(v, __shfl_xor(v, 4, 32));
      v = fmaxf(v, __shfl_xor(v, 8, 32));
      mnew[i] = fmaxf(mrun[i], v);
      scl[i]  = __expf(mrun[i] - mnew[i]);
      mrun[i] = mnew[i];
    }
    #pragma unroll
    for (int i = 0; i < 8; ++i) {
      float p0 = __expf(sc[0][i] - mnew[i]);
      float p1 = __expf(sc[1][i] - mnew[i]);
      sc[0][i] = p0; sc[1][i] = p1;
      float r = p0 + p1;
      r += __shfl_xor(r, 1, 32);
      r += __shfl_xor(r, 2, 32);
      r += __shfl_xor(r, 4, 32);
      r += __shfl_xor(r, 8, 32);
      lrun[i] = lrun[i] * scl[i] + r;
    }
    #pragma unroll
    for (int t = 0; t < 8; ++t)
      #pragma unroll
      for (int i = 0; i < 8; ++i) acc[t][i] *= scl[i];

    // ---- P: C-layout -> A-layout via LDS (private per wave) ----
    __builtin_amdgcn_wave_barrier();
    #pragma unroll
    for (int t = 0; t < 2; ++t)
      #pragma unroll
      for (int i = 0; i < 8; ++i)
        pls[wave][i + 8 * hl][t * 16 + nn] = f2bf(sc[t][i]);
    __builtin_amdgcn_wave_barrier();
    asm volatile("s_wait_dscnt 0" ::: "memory");
    v16bf pf;
    {
      const unsigned short* p = &pls[wave][lane & 15][hl * 8];
      union { v16bf v; v8bf h2[2]; } u;
      u.h2[0] = *(const v8bf*)p;
      u.h2[1] = *(const v8bf*)(p + 16);
      pf = u.v;
    }
    __builtin_amdgcn_wave_barrier();

    // ---- O += P V  (V row-major is already B-layout friendly) ----
    #pragma unroll
    for (int t = 0; t < 8; ++t) {
      v16bf vf = load_b_frag(Vp, HD_, kv0, t * 16, lane);
      acc[t] = __builtin_amdgcn_wmma_f32_16x16x32_bf16(
          false, pf, false, vf, (short)0, acc[t], false, false);
    }
  }

  // ---- normalize and store O as [b, s, h*HD + d] bf16 ----
  float invl[8];
  #pragma unroll
  for (int i = 0; i < 8; ++i) invl[i] = 1.0f / lrun[i];
  #pragma unroll
  for (int t = 0; t < 8; ++t)
    #pragma unroll
    for (int i = 0; i < 8; ++i) {
      int qq = qbase + i + 8 * hl;
      Ob[((size_t)(bb * S_) + qq) * D_ + hh * HD_ + t * 16 + nn] =
          f2bf(acc[t][i] * invl[i]);
    }
}

// ---------------- output projection: O[4096][2048] @ Wout[2048][2048] -> f32 ----
__global__ void __launch_bounds__(256) gemm_out_kernel(
    const unsigned short* __restrict__ A,
    const unsigned short* __restrict__ W,
    float* __restrict__ Cout) {
  __shared__ __align__(16) unsigned short wtile[2][32][512];  // 64 KB, LDS off 0
  int lane = threadIdx.x & 31;
  int wave = threadIdx.x >> 5;
  int nblk0 = blockIdx.x * 512;
  int row0  = blockIdx.y * 16;

  v8f acc[4];
  gemm_core(A, D_, W, D_, D_, row0, nblk0, &wtile[0][0][0], lane, wave, acc);

  int hl = lane >> 4, nn = lane & 15;
  int r0 = row0 + 8 * hl;
  int n0 = nblk0 + wave * 64;
  #pragma unroll
  for (int j = 0; j < 4; ++j)
    #pragma unroll
    for (int i = 0; i < 8; ++i)
      Cout[(size_t)(r0 + i) * D_ + n0 + j * 16 + nn] = acc[j][i];
}

extern "C" void kernel_launch(void* const* d_in, const int* in_sizes, int n_in,
                              void* d_out, int out_size, void* d_ws, size_t ws_size,
                              hipStream_t stream) {
  const float* x    = (const float*)d_in[0];
  const float* Wqkv = (const float*)d_in[1];
  const float* Wout = (const float*)d_in[2];
  float* out = (float*)d_out;

  unsigned short* ws = (unsigned short*)d_ws;
  size_t off = 0;
  unsigned short* xb    = ws + off; off += (size_t)MROWS_ * D_;
  unsigned short* wqkvb = ws + off; off += (size_t)D_ * NQKV_;
  unsigned short* woutb = ws + off; off += (size_t)D_ * D_;
  unsigned short* Qb    = ws + off; off += (size_t)B_ * H_ * S_ * HD_;
  unsigned short* Ktb   = ws + off; off += (size_t)B_ * H_ * S_ * HD_;
  unsigned short* Vb    = ws + off; off += (size_t)B_ * H_ * S_ * HD_;
  unsigned short* Ob    = ws + off; off += (size_t)MROWS_ * D_;
  (void)ws_size; (void)in_sizes; (void)n_in; (void)out_size;

  cvt_bf16_kernel<<<1024, 256, 0, stream>>>(x,    xb,    MROWS_ * D_);
  cvt_bf16_kernel<<<1024, 256, 0, stream>>>(Wqkv, wqkvb, D_ * NQKV_);
  cvt_bf16_kernel<<<1024, 256, 0, stream>>>(Wout, woutb, D_ * D_);

  gemm_qkv_kernel<<<dim3(NQKV_ / 512, MROWS_ / 16), 256, 0, stream>>>(
      xb, wqkvb, Qb, Ktb, Vb);

  attn_kernel<<<dim3(S_ / 64, H_, B_), 128, 0, stream>>>(Qb, Ktb, Vb, Ob);

  gemm_out_kernel<<<dim3(D_ / 512, MROWS_ / 16), 256, 0, stream>>>(
      Ob, woutb, out);
}
